// myGRU_4552665334192
// MI455X (gfx1250) — matrix operand order repaired and checked by reading, not compiled, not run on hardware
//
#include <hip/hip_runtime.h>

typedef unsigned short u16;
typedef __attribute__((ext_vector_type(16))) __bf16 v16bf;
typedef __attribute__((ext_vector_type(8)))  float  v8f;

#define WMMA_BF16(C, A, Bm)                                                    \
  (C) = __builtin_amdgcn_wmma_f32_16x16x32_bf16(false, (A), false, (Bm),       \
                                                (short)0, (C), false, false)

__device__ __forceinline__ u16 f2bf(float x) {
  unsigned u = __float_as_uint(x);
  unsigned r = u + 0x7FFFu + ((u >> 16) & 1u);  // round-to-nearest-even
  return (u16)(r >> 16);
}
__device__ __forceinline__ float bf2f(u16 h) {
  return __uint_as_float(((unsigned)h) << 16);
}
// Fast transcendental gates: v_exp_f32 + v_rcp_f32 instead of libm tanhf /
// IEEE division (keeps the serial gate phase off the critical-path budget).
__device__ __forceinline__ float fast_rcp(float x) {
  return __builtin_amdgcn_rcpf(x);
}
__device__ __forceinline__ float sigmoidf_(float x) {
  return fast_rcp(1.0f + __expf(-x));
}
__device__ __forceinline__ float tanhf_(float x) {
  // tanh(x) = 1 - 2/(exp(2x)+1); saturates correctly to +/-1 for large |x|
  return 1.0f - 2.0f * fast_rcp(1.0f + __expf(2.0f * x));
}

union FragU { v16bf v; uint4 q[2]; };

// A fragment (16x32 bf16): lane m<16 -> row m, K {kb..kb+7, kb+16..kb+23};
// lane m+16 -> row m, K {kb+8..kb+15, kb+24..kb+31}.
__device__ __forceinline__ v16bf load_fragA(const u16* base, int row, int stride,
                                            int kb, int lane) {
  FragU f;
  const u16* p = base + row * stride + kb + ((lane >> 4) << 3);
  f.q[0] = *(const uint4*)(p);
  f.q[1] = *(const uint4*)(p + 16);
  return f.v;
}
// B fragment (32x16 bf16): lane n<16 -> col n, K {kb..kb+7, kb+16..kb+23}; lane n+16 -> K {kb+8..kb+15, kb+24..kb+31}.
// Weight array is row-major [3H][K] so "col" = weight row g, contiguous in K.
__device__ __forceinline__ v16bf load_fragB(const u16* base, int row, int stride,
                                            int kb, int lane) {
  FragU f;
  const u16* p = base + row * stride + kb + ((lane >> 4) << 3);
  f.q[0] = *(const uint4*)(p);
  f.q[1] = *(const uint4*)(p + 16);
  return f.v;
}

// Accumulate A[32xK] * W^T into r/z/n accumulators for this wave's unit slice.
// accR/accZ/accN are [2*NT] v8f (mt-major). bf16x3: hi*hi + hi*lo + lo*hi.
template <int K, int H, int NT>
__device__ __forceinline__ void gemm3(const u16* aHiM, const u16* aLoM,
                                      const u16* wHi, const u16* wLo,
                                      v8f* accR, v8f* accZ, v8f* accN,
                                      int jbase, int col, int lane) {
#pragma unroll 1
  for (int kb = 0; kb < K; kb += 32) {
    v16bf a0h = load_fragA(aHiM, col,      K, kb, lane);
    v16bf a0l = load_fragA(aLoM, col,      K, kb, lane);
    v16bf a1h = load_fragA(aHiM, col + 16, K, kb, lane);
    v16bf a1l = load_fragA(aLoM, col + 16, K, kb, lane);
#pragma unroll
    for (int nt = 0; nt < NT; ++nt) {
#pragma unroll
      for (int g = 0; g < 3; ++g) {
        const int wrow = g * H + jbase + nt * 16 + col;
        v16bf bh = load_fragB(wHi, wrow, K, kb, lane);
        v16bf bl = load_fragB(wLo, wrow, K, kb, lane);
        v8f* tgt = (g == 0) ? accR : (g == 1) ? accZ : accN;
        WMMA_BF16(tgt[nt],      a0h, bh);
        WMMA_BF16(tgt[nt],      a0h, bl);
        WMMA_BF16(tgt[nt],      a0l, bh);
        WMMA_BF16(tgt[NT + nt], a1h, bh);
        WMMA_BF16(tgt[NT + nt], a1h, bl);
        WMMA_BF16(tgt[NT + nt], a1l, bh);
      }
    }
  }
}

// One timestep of one GRU layer for a 32-row batch tile.
// hHi/hLo: LDS bf16 hi/lo state [32][H]; hReg: f32 state in C-tile layout (regs).
template <bool SCALAR_IN, int KIN, int H, int NT>
__device__ __forceinline__ void layer_step(
    const float* xs, const float* wih_f32,           // layer-1 scalar-input path
    const u16* pHi, const u16* pLo,                  // prev layer h (A for input GEMM)
    const u16* wihHi, const u16* wihLo,
    const u16* whhHi, const u16* whhLo,
    const float* bih, const float* bhh,
    u16* hHi, u16* hLo, v8f* hReg, int wave, int lane) {
  const int col   = lane & 15;
  const int khalf = (lane >> 4) << 3;    // C-tile row offset: 0 or 8
  const int jbase = wave * (H / 8);      // this wave's unit slice

  v8f aR[2 * NT], aZ[2 * NT], aNX[2 * NT], aNH[2 * NT];
#pragma unroll
  for (int mt = 0; mt < 2; ++mt) {
#pragma unroll
    for (int nt = 0; nt < NT; ++nt) {
      const int idx = mt * NT + nt;
      const int j   = jbase + nt * 16 + col;
      const float br  = bih[j] + bhh[j];
      const float bz  = bih[H + j] + bhh[H + j];
      const float bnx = bih[2 * H + j];
      const float bnh = bhh[2 * H + j];
      v8f vr, vz, vnx, vnh;
#pragma unroll
      for (int i = 0; i < 8; ++i) { vr[i] = br; vz[i] = bz; vnx[i] = bnx; vnh[i] = bnh; }
      if constexpr (SCALAR_IN) {  // layer 1: input size 1 -> outer product
        const float w0 = wih_f32[j], w1 = wih_f32[H + j], w2 = wih_f32[2 * H + j];
#pragma unroll
        for (int i = 0; i < 8; ++i) {
          const float xv = xs[mt * 16 + khalf + i];
          vr[i] += xv * w0; vz[i] += xv * w1; vnx[i] += xv * w2;
        }
      }
      aR[idx] = vr; aZ[idx] = vz; aNX[idx] = vnx; aNH[idx] = vnh;
    }
  }

  if constexpr (!SCALAR_IN)
    gemm3<KIN, H, NT>(pHi, pLo, wihHi, wihLo, aR, aZ, aNX, jbase, col, lane);
  gemm3<H, H, NT>(hHi, hLo, whhHi, whhLo, aR, aZ, aNH, jbase, col, lane);

  __syncthreads();  // all waves done reading hHi/hLo before overwrite

#pragma unroll
  for (int mt = 0; mt < 2; ++mt) {
#pragma unroll
    for (int nt = 0; nt < NT; ++nt) {
      const int idx = mt * NT + nt;
      const int j   = jbase + nt * 16 + col;
      v8f hold = hReg[idx];
      v8f hnew;
#pragma unroll
      for (int i = 0; i < 8; ++i) {
        const float r = sigmoidf_(aR[idx][i]);
        const float z = sigmoidf_(aZ[idx][i]);
        const float n = tanhf_(aNX[idx][i] + r * aNH[idx][i]);
        hnew[i] = (1.0f - z) * n + z * hold[i];
      }
      hReg[idx] = hnew;
#pragma unroll
      for (int i = 0; i < 8; ++i) {
        const int row = mt * 16 + khalf + i;
        const float v = hnew[i];
        const u16 hi  = f2bf(v);
        hHi[row * H + j] = hi;
        hLo[row * H + j] = f2bf(v - bf2f(hi));
      }
    }
  }
  __syncthreads();  // state visible to next consumer
}

struct GruParams {
  const float* x;
  const float* wih1; const float* bih1; const float* bhh1;
  const float* bih2; const float* bhh2;
  const float* bih3; const float* bhh3;
  const float* bih4; const float* bhh4;
  const float* w_out; const float* b_out;
  const u16 *whh1Hi, *whh1Lo;
  const u16 *wih2Hi, *wih2Lo, *whh2Hi, *whh2Lo;
  const u16 *wih3Hi, *wih3Lo, *whh3Hi, *whh3Lo;
  const u16 *wih4Hi, *wih4Lo, *whh4Hi, *whh4Lo;
  float* out;
};

#define T_STEPS 1024

__global__ __launch_bounds__(256, 1) void gru_fused(GruParams p) {
  extern __shared__ char smem[];
  u16* h1Hi = (u16*)smem;            // [32][256]
  u16* h1Lo = h1Hi + 32 * 256;
  u16* h4Hi = h1Lo + 32 * 256;       // [32][256]
  u16* h4Lo = h4Hi + 32 * 256;
  u16* h2Hi = h4Lo + 32 * 256;       // [32][128]
  u16* h2Lo = h2Hi + 32 * 128;
  u16* h3Hi = h2Lo + 32 * 128;
  u16* h3Lo = h3Hi + 32 * 128;
  float* xs   = (float*)(h3Lo + 32 * 128);  // [32]
  float* yred = xs + 32;                     // [8][32]

  const int tid  = threadIdx.x;
  const int wave = tid >> 5;
  const int lane = tid & 31;
  const int b0   = blockIdx.x * 32;

  for (int i = tid; i < 32 * 2 * (256 + 128 + 128 + 256); i += 256) h1Hi[i] = 0;
  v8f hR1[4], hR2[2], hR3[2], hR4[4];
#pragma unroll
  for (int i = 0; i < 4; ++i) { v8f z = {}; hR1[i] = z; hR4[i] = z; }
#pragma unroll
  for (int i = 0; i < 2; ++i) { v8f z = {}; hR2[i] = z; hR3[i] = z; }
  __syncthreads();

  for (int t = 0; t < T_STEPS; ++t) {
    if (tid < 32) xs[tid] = p.x[(b0 + tid) * T_STEPS + t];
    __syncthreads();

    layer_step<true, 32, 256, 2>(xs, p.wih1, nullptr, nullptr, nullptr, nullptr,
                                 p.whh1Hi, p.whh1Lo, p.bih1, p.bhh1,
                                 h1Hi, h1Lo, hR1, wave, lane);
    layer_step<false, 256, 128, 1>(nullptr, nullptr, h1Hi, h1Lo,
                                   p.wih2Hi, p.wih2Lo, p.whh2Hi, p.whh2Lo,
                                   p.bih2, p.bhh2, h2Hi, h2Lo, hR2, wave, lane);
    layer_step<false, 128, 128, 1>(nullptr, nullptr, h2Hi, h2Lo,
                                   p.wih3Hi, p.wih3Lo, p.whh3Hi, p.whh3Lo,
                                   p.bih3, p.bhh3, h3Hi, h3Lo, hR3, wave, lane);
    layer_step<false, 128, 256, 2>(nullptr, nullptr, h3Hi, h3Lo,
                                   p.wih4Hi, p.wih4Lo, p.whh4Hi, p.whh4Lo,
                                   p.bih4, p.bhh4, h4Hi, h4Lo, hR4, wave, lane);

    // y[b,t] = h4[b,:] . w_out + b_out
    const int row = tid & 31, chunk = tid >> 5;
    float s = 0.0f;
#pragma unroll
    for (int u = 0; u < 32; ++u) {
      const int uu = chunk * 32 + u;
      s += (bf2f(h4Hi[row * 256 + uu]) + bf2f(h4Lo[row * 256 + uu])) * p.w_out[uu];
    }
    yred[chunk * 32 + row] = s;
    __syncthreads();
    if (tid < 32) {
      float y = p.b_out[0];
#pragma unroll
      for (int c = 0; c < 8; ++c) y += yred[c * 32 + tid];
      p.out[(b0 + tid) * T_STEPS + t] = y;
    }
    __syncthreads();
  }
}

__global__ void split_w(const float* __restrict__ src, u16* __restrict__ hi,
                        u16* __restrict__ lo, int n) {
  int i = blockIdx.x * 256 + threadIdx.x;
  if (i < n) {
    float x = src[i];
    u16 h = f2bf(x);
    hi[i] = h;
    lo[i] = f2bf(x - bf2f(h));
  }
}

extern "C" void kernel_launch(void* const* d_in, const int* in_sizes, int n_in,
                              void* d_out, int out_size, void* d_ws, size_t ws_size,
                              hipStream_t stream) {
  (void)in_sizes; (void)n_in; (void)out_size; (void)ws_size;
  const float* x    = (const float*)d_in[0];
  const float* wih1 = (const float*)d_in[1];
  const float* whh1 = (const float*)d_in[2];
  const float* bih1 = (const float*)d_in[3];
  const float* bhh1 = (const float*)d_in[4];
  const float* wih2 = (const float*)d_in[5];
  const float* whh2 = (const float*)d_in[6];
  const float* bih2 = (const float*)d_in[7];
  const float* bhh2 = (const float*)d_in[8];
  const float* wih3 = (const float*)d_in[9];
  const float* whh3 = (const float*)d_in[10];
  const float* bih3 = (const float*)d_in[11];
  const float* bhh3 = (const float*)d_in[12];
  const float* wih4 = (const float*)d_in[13];
  const float* whh4 = (const float*)d_in[14];
  const float* bih4 = (const float*)d_in[15];
  const float* bhh4 = (const float*)d_in[16];
  const float* w_out = (const float*)d_in[17];
  const float* b_out = (const float*)d_in[18];

  u16* ws = (u16*)d_ws;
  size_t off = 0;
  auto take2 = [&](int n, u16** hi, u16** lo) {
    *hi = ws + off; off += (size_t)n;
    *lo = ws + off; off += (size_t)n;
  };
  u16 *whh1Hi, *whh1Lo, *wih2Hi, *wih2Lo, *whh2Hi, *whh2Lo;
  u16 *wih3Hi, *wih3Lo, *whh3Hi, *whh3Lo, *wih4Hi, *wih4Lo, *whh4Hi, *whh4Lo;
  take2(768 * 256, &whh1Hi, &whh1Lo);
  take2(384 * 256, &wih2Hi, &wih2Lo);
  take2(384 * 128, &whh2Hi, &whh2Lo);
  take2(384 * 128, &wih3Hi, &wih3Lo);
  take2(384 * 128, &whh3Hi, &whh3Lo);
  take2(768 * 128, &wih4Hi, &wih4Lo);
  take2(768 * 256, &whh4Hi, &whh4Lo);

  auto conv = [&](const float* s, u16* hi, u16* lo, int n) {
    split_w<<<(n + 255) / 256, 256, 0, stream>>>(s, hi, lo, n);
  };
  conv(whh1, whh1Hi, whh1Lo, 768 * 256);
  conv(wih2, wih2Hi, wih2Lo, 384 * 256);
  conv(whh2, whh2Hi, whh2Lo, 384 * 128);
  conv(wih3, wih3Hi, wih3Lo, 384 * 128);
  conv(whh3, whh3Hi, whh3Lo, 384 * 128);
  conv(wih4, wih4Hi, wih4Lo, 768 * 128);
  conv(whh4, whh4Hi, whh4Lo, 768 * 256);

  GruParams p;
  p.x = x; p.wih1 = wih1; p.bih1 = bih1; p.bhh1 = bhh1;
  p.bih2 = bih2; p.bhh2 = bhh2; p.bih3 = bih3; p.bhh3 = bhh3;
  p.bih4 = bih4; p.bhh4 = bhh4; p.w_out = w_out; p.b_out = b_out;
  p.whh1Hi = whh1Hi; p.whh1Lo = whh1Lo;
  p.wih2Hi = wih2Hi; p.wih2Lo = wih2Lo; p.whh2Hi = whh2Hi; p.whh2Lo = whh2Lo;
  p.wih3Hi = wih3Hi; p.wih3Lo = wih3Lo; p.whh3Hi = whh3Hi; p.whh3Lo = whh3Lo;
  p.wih4Hi = wih4Hi; p.wih4Lo = wih4Lo; p.whh4Hi = whh4Hi; p.whh4Lo = whh4Lo;
  p.out = (float*)d_out;

  const int SMEM = 32 * 2 * (256 + 128 + 128 + 256) * 2 + (32 + 8 * 32) * 4;  // 99456 B
  (void)hipFuncSetAttribute((const void*)gru_fused,
                            hipFuncAttributeMaxDynamicSharedMemorySize, SMEM);
  gru_fused<<<8, 256, SMEM, stream>>>(p);
}
